// SelfAttention_34119220200031
// MI455X (gfx1250) — matrix operand order repaired and checked by reading, not compiled, not run on hardware
//
#include <hip/hip_runtime.h>
#include <hip/hip_bf16.h>

// ---------------------------------------------------------------------------
// SAGAN self-attention, fused flash-attention style for gfx1250 (MI455X).
// Transposed-tile formulation: S^T = phi * theta, O^T = G^T * P^T so that a
// query maps to a lane -> online softmax needs only 1 cross-lane op per
// reduction (vs 32 ds_bpermute butterflies in the row-major formulation).
// ---------------------------------------------------------------------------

typedef __attribute__((ext_vector_type(2)))  float  v2f;
typedef __attribute__((ext_vector_type(8)))  float  v8f;
typedef __attribute__((ext_vector_type(8)))  __bf16 v8bf;
typedef __attribute__((ext_vector_type(16))) __bf16 v16bf;
typedef __attribute__((ext_vector_type(4)))  int    v4i;

#define C_IN   64
#define NHW    4096
#define KP     1024     // pooled key count
#define CTH    8        // theta/phi channels (C/8)
#define CG     32       // g channels (C/2)
#define KC     32       // key chunk for flash loop
#define NCHUNK (KP / KC)
#define WAVES  8
#define QTILE  16
#define QBLK   (WAVES * QTILE)   // 128 queries per workgroup
#define PSTR   40                // padded bf16 row stride for P/O bounce
#define OPADQ  17                // padded float row stride for out bounce

static __device__ __forceinline__ void wave_lds_fence() {
  asm volatile("s_wait_dscnt 0x0" ::: "memory");
}

// ---- optional CDNA5 async global->LDS staging -----------------------------
#if defined(__HIP_DEVICE_COMPILE__) && \
    __has_builtin(__builtin_amdgcn_global_load_async_to_lds_b128)
#define HAVE_ASYNC_LDS 1
#else
#define HAVE_ASYNC_LDS 0
#endif

#if HAVE_ASYNC_LDS
typedef __attribute__((address_space(1))) v4i GV4;   // global int4
typedef __attribute__((address_space(3))) v4i LV4;   // LDS int4
static __device__ __forceinline__ void async_cp16(void* lds, const void* gmem) {
  __builtin_amdgcn_global_load_async_to_lds_b128((GV4*)gmem, (LV4*)lds, 0, 0);
}
static __device__ __forceinline__ void wait_async0() {
#if __has_builtin(__builtin_amdgcn_s_wait_asynccnt)
  __builtin_amdgcn_s_wait_asynccnt(0);
#else
  asm volatile("s_wait_asynccnt 0x0" ::: "memory");
#endif
}
#endif

// ---------------------------------------------------------------------------
// Kernel 1: theta projection.  theta[b][q][c'] (8 contiguous per query).
// ---------------------------------------------------------------------------
__global__ __launch_bounds__(256) void theta_proj_kernel(
    const float* __restrict__ x, const float* __restrict__ Wt,
    float* __restrict__ theta) {
  int idx = blockIdx.x * 256 + threadIdx.x;   // [0, B*HW)
  int b = idx >> 12;
  int q = idx & (NHW - 1);
  const float* xb = x + (size_t)b * C_IN * NHW + q;
  float acc[CTH];
#pragma unroll
  for (int o = 0; o < CTH; ++o) acc[o] = 0.f;
#pragma unroll 8
  for (int c = 0; c < C_IN; ++c) {
    float xv = xb[(size_t)c * NHW];
#pragma unroll
    for (int o = 0; o < CTH; ++o) acc[o] = fmaf(Wt[o * C_IN + c], xv, acc[o]);
  }
  float* t = theta + (size_t)idx * CTH;
#pragma unroll
  for (int o = 0; o < CTH; ++o) t[o] = acc[o];
}

// ---------------------------------------------------------------------------
// Kernel 2: phi & g projections fused with 2x2 maxpool, channel-major.
// ---------------------------------------------------------------------------
__global__ __launch_bounds__(256) void pool_proj_kernel(
    const float* __restrict__ x, const float* __restrict__ Wphi,
    const float* __restrict__ Wg, float* __restrict__ phi,
    __bf16* __restrict__ g) {
  int idx = blockIdx.x * 256 + threadIdx.x;   // [0, B*KP)
  int b  = idx >> 10;
  int kp = idx & (KP - 1);
  int hp = kp >> 5, wp = kp & 31;
  int q00 = (hp * 2) * 64 + wp * 2;
  const float* xb = x + (size_t)b * C_IN * NHW;

  float pbest[CTH], gbest[CG];
#pragma unroll
  for (int o = 0; o < CTH; ++o) pbest[o] = -3.0e38f;
#pragma unroll
  for (int o = 0; o < CG; ++o) gbest[o] = -3.0e38f;

  const int off[4] = {0, 1, 64, 65};
#pragma unroll
  for (int p = 0; p < 4; ++p) {
    int qp = q00 + off[p];
    float pa[CTH], ga[CG];
#pragma unroll
    for (int o = 0; o < CTH; ++o) pa[o] = 0.f;
#pragma unroll
    for (int o = 0; o < CG; ++o) ga[o] = 0.f;
#pragma unroll 4
    for (int c = 0; c < C_IN; ++c) {
      float xv = xb[(size_t)c * NHW + qp];
#pragma unroll
      for (int o = 0; o < CTH; ++o) pa[o] = fmaf(Wphi[o * C_IN + c], xv, pa[o]);
#pragma unroll
      for (int o = 0; o < CG; ++o) ga[o] = fmaf(Wg[o * C_IN + c], xv, ga[o]);
    }
#pragma unroll
    for (int o = 0; o < CTH; ++o) pbest[o] = fmaxf(pbest[o], pa[o]);
#pragma unroll
    for (int o = 0; o < CG; ++o) gbest[o] = fmaxf(gbest[o], ga[o]);
  }
  float*  pd = phi + (size_t)b * CTH * KP + kp;
  __bf16* gd = g   + (size_t)b * CG  * KP + kp;
#pragma unroll
  for (int o = 0; o < CTH; ++o) pd[(size_t)o * KP] = pbest[o];
#pragma unroll
  for (int o = 0; o < CG; ++o) gd[(size_t)o * KP] = (__bf16)gbest[o];
}

// ---------------------------------------------------------------------------
// Kernel 3: fused flash attention (transposed tiles) + projection + residual.
// ---------------------------------------------------------------------------
__global__ __launch_bounds__(256) void attn_kernel(
    const float* __restrict__ x, const float* __restrict__ theta,
    const float* __restrict__ phi, const __bf16* __restrict__ g,
    const float* __restrict__ Wattn, const float* __restrict__ sigma_p,
    float* __restrict__ out) {
  __shared__ __align__(16) float  lphi[CTH * KP];             // 32 KB
  __shared__ __align__(16) __bf16 lg[CG * KP];                // 64 KB
  __shared__ __align__(16) __bf16 lPb[WAVES][QTILE * PSTR];   // 10 KB
  __shared__ __align__(16) float  lOut[WAVES][C_IN * OPADQ];  // 34 KB

  int b  = blockIdx.x >> 5;
  int qb = (blockIdx.x & 31) * QBLK;
  int tid = threadIdx.x;
  int w = tid >> 5;          // wave id
  int l = tid & 31;          // lane
  int n = l & 15;            // query column for this lane (S^T/O^T tiles)
  int h = l >> 4;            // lane half
  int hb = h * 8;            // A-layout K base for VGPRs 0..3

  // ---- stage phi (f32) and g (bf16) for this batch into LDS ---------------
#if HAVE_ASYNC_LDS
  {
    const char* ps = (const char*)(phi + (size_t)b * CTH * KP);
    char* pd = (char*)lphi;
    for (int i = tid; i < (CTH * KP * 4) / 16; i += 256)
      async_cp16(pd + (size_t)i * 16, ps + (size_t)i * 16);
    const char* gs = (const char*)(g + (size_t)b * CG * KP);
    char* gd = (char*)lg;
    for (int i = tid; i < (CG * KP * 2) / 16; i += 256)
      async_cp16(gd + (size_t)i * 16, gs + (size_t)i * 16);
    wait_async0();
  }
#else
  {
    const float4* ps = (const float4*)(phi + (size_t)b * CTH * KP);
    float4* pd = (float4*)lphi;
    for (int i = tid; i < (CTH * KP) / 4; i += 256) pd[i] = ps[i];
    const float4* gs = (const float4*)(g + (size_t)b * CG * KP);
    float4* gd = (float4*)lg;
    for (int i = tid; i < (CG * KP) / 8; i += 256) gd[i] = gs[i];
  }
#endif
  __syncthreads();

  // ---- loop-invariant theta B operand (f32 B-matrix 4x16, 2 K-chunks) ------
  // B[c'][q]: VGPR0 = row 2h, VGPR1 = row 2h+1 at column n.
  const float* tq = theta + ((size_t)b * NHW + qb + w * QTILE + n) * CTH;
  v2f bq0 = { tq[2 * h],     tq[2 * h + 1] };      // c' 0..3
  v2f bq1 = { tq[2 * h + 4], tq[2 * h + 5] };      // c' 4..7

  // ---- flash state: one query per lane (replicated across lane halves) -----
  float m_run = -3.0e38f, l_run = 0.f;
  v8f ot0 = {}, ot1 = {};   // O^T tiles: rows = g-channels (v+8h), col = query n
  __bf16* pb = &lPb[w][0];

  for (int kc = 0; kc < NCHUNK; ++kc) {
    int kb = kc * KC;

    // S^T tiles [16 keys x 16 queries] via f32 WMMA 16x16x4.
    // A[m=key][K=c']: VGPR0 = c'=2h, VGPR1 = c'=2h+1 at key row m.
    v8f st0 = {}, st1 = {};
    {
      int k0 = kb + n;           // tile0 key row for this lane
      v2f a00 = { lphi[(size_t)(2 * h + 0) * KP + k0],
                  lphi[(size_t)(2 * h + 1) * KP + k0] };
      v2f a01 = { lphi[(size_t)(2 * h + 4) * KP + k0],
                  lphi[(size_t)(2 * h + 5) * KP + k0] };
      st0 = __builtin_amdgcn_wmma_f32_16x16x4_f32(false, a00, false, bq0,
                                                  (short)0, st0, false, false);
      st0 = __builtin_amdgcn_wmma_f32_16x16x4_f32(false, a01, false, bq1,
                                                  (short)0, st0, false, false);
      int k1 = k0 + 16;          // tile1 key row
      v2f a10 = { lphi[(size_t)(2 * h + 0) * KP + k1],
                  lphi[(size_t)(2 * h + 1) * KP + k1] };
      v2f a11 = { lphi[(size_t)(2 * h + 4) * KP + k1],
                  lphi[(size_t)(2 * h + 5) * KP + k1] };
      st1 = __builtin_amdgcn_wmma_f32_16x16x4_f32(false, a10, false, bq0,
                                                  (short)0, st1, false, false);
      st1 = __builtin_amdgcn_wmma_f32_16x16x4_f32(false, a11, false, bq1,
                                                  (short)0, st1, false, false);
    }

    // ---- online softmax: all 16 values in this lane belong to query n ------
    float mx;
    {
      float t[8];
#pragma unroll
      for (int v = 0; v < 8; ++v) t[v] = fmaxf(st0[v], st1[v]);
#pragma unroll
      for (int s = 4; s > 0; s >>= 1)
#pragma unroll
        for (int v = 0; v < 4; ++v)
          if (v < s) t[v] = fmaxf(t[v], t[v + s]);
      mx = fmaxf(t[0], __shfl_xor(t[0], 16, 32));  // combine lane halves
    }
    float mnew  = fmaxf(m_run, mx);
    float scale = __expf(m_run - mnew);
    float p0[8], p1[8];
#pragma unroll
    for (int v = 0; v < 8; ++v) {
      p0[v] = __expf(st0[v] - mnew);
      p1[v] = __expf(st1[v] - mnew);
    }
    {
      float t[8];
#pragma unroll
      for (int v = 0; v < 8; ++v) t[v] = p0[v] + p1[v];
#pragma unroll
      for (int s = 4; s > 0; s >>= 1)
#pragma unroll
        for (int v = 0; v < 4; ++v)
          if (v < s) t[v] += t[v + s];
      float rs = t[0] + __shfl_xor(t[0], 16, 32);
      l_run = l_run * scale + rs;
    }
    m_run = mnew;
#pragma unroll
    for (int v = 0; v < 8; ++v) { ot0[v] *= scale; ot1[v] *= scale; }

    // ---- bounce P^T to LDS as [q][k] bf16, packed b128 stores --------------
    {
      v8bf pk0, pk1;
#pragma unroll
      for (int v = 0; v < 8; ++v) { pk0[v] = (__bf16)p0[v]; pk1[v] = (__bf16)p1[v]; }
      *(v8bf*)(pb + n * PSTR + 8 * h)      = pk0;   // keys kb + 8h + v (tile0)
      *(v8bf*)(pb + n * PSTR + 16 + 8 * h) = pk1;   // keys kb+16+8h+v (tile1)
    }
    wave_lds_fence();

    // B = P^T [32 keys x 16 q]: lane col n, keys 8h..8h+7 and 16+8h..16+8h+7 per half.
    v16bf pB;
    {
      const v8bf* sp = (const v8bf*)(pb + n * PSTR + 8 * h);
      v8bf lo = sp[0], hi = sp[2];
      pB = __builtin_shufflevector(lo, hi, 0, 1, 2, 3, 4, 5, 6, 7,
                                   8, 9, 10, 11, 12, 13, 14, 15);
    }

    // A = G^T [16 c x 32 k] bf16 A-layout: contiguous k in channel-major lg.
#pragma unroll
    for (int t = 0; t < 2; ++t) {
      const __bf16* gr = lg + (size_t)(t * 16 + n) * KP + kb;
      v8bf lo = *(const v8bf*)(gr + hb);
      v8bf hi = *(const v8bf*)(gr + hb + 16);
      v16bf ga = __builtin_shufflevector(lo, hi, 0, 1, 2, 3, 4, 5, 6, 7,
                                         8, 9, 10, 11, 12, 13, 14, 15);
      if (t == 0)
        ot0 = __builtin_amdgcn_wmma_f32_16x16x32_bf16(false, ga, false, pB,
                                                      (short)0, ot0, false, false);
      else
        ot1 = __builtin_amdgcn_wmma_f32_16x16x32_bf16(false, ga, false, pB,
                                                      (short)0, ot1, false, false);
    }
  }

  // ---- normalize (per-lane scalar!) and bounce O^T as [q][c] ----------------
  {
    float inv = 1.0f / l_run;
    v8bf ok0, ok1;
#pragma unroll
    for (int v = 0; v < 8; ++v) {
      ok0[v] = (__bf16)(ot0[v] * inv);
      ok1[v] = (__bf16)(ot1[v] * inv);
    }
    *(v8bf*)(pb + n * PSTR + 8 * h)      = ok0;   // channels 8h+v
    *(v8bf*)(pb + n * PSTR + 16 + 8 * h) = ok1;   // channels 16+8h+v
  }
  wave_lds_fence();

  // B = O^T [32 c x 16 q] B-layout
  v16bf oB;
  {
    const v8bf* sp = (const v8bf*)(pb + n * PSTR + 8 * h);
    v8bf lo = sp[0], hi = sp[2];
    oB = __builtin_shufflevector(lo, hi, 0, 1, 2, 3, 4, 5, 6, 7,
                                 8, 9, 10, 11, 12, 13, 14, 15);
  }

  // ---- out^T = W_attn * O^T : 4 M-tiles of output channels -----------------
  float* ob = &lOut[w][0];
#pragma unroll
  for (int t = 0; t < 4; ++t) {
    int o = t * 16 + n;                       // A row = output channel
    const float4* w0 = (const float4*)(Wattn + (size_t)o * CG + hb);
    const float4* w1 = (const float4*)(Wattn + (size_t)o * CG + hb + 16);
    float4 wa = w0[0], wbv = w0[1], wc = w1[0], wd = w1[1];
    v16bf wA;
    wA[0] = (__bf16)wa.x;  wA[1] = (__bf16)wa.y;
    wA[2] = (__bf16)wa.z;  wA[3] = (__bf16)wa.w;
    wA[4] = (__bf16)wbv.x; wA[5] = (__bf16)wbv.y;
    wA[6] = (__bf16)wbv.z; wA[7] = (__bf16)wbv.w;
    wA[8]  = (__bf16)wc.x; wA[9]  = (__bf16)wc.y;
    wA[10] = (__bf16)wc.z; wA[11] = (__bf16)wc.w;
    wA[12] = (__bf16)wd.x; wA[13] = (__bf16)wd.y;
    wA[14] = (__bf16)wd.z; wA[15] = (__bf16)wd.w;
    v8f r = {};
    r = __builtin_amdgcn_wmma_f32_16x16x32_bf16(false, wA, false, oB,
                                                (short)0, r, false, false);
#pragma unroll
    for (int v = 0; v < 8; ++v)
      ob[(t * 16 + v + 8 * h) * OPADQ + n] = r[v];   // lOut[o][q]
  }
  wave_lds_fence();

  // ---- residual + sigma scale, coalesced stores ----------------------------
  float sg = sigma_p[0];
  int qwave = qb + w * QTILE;
  const float* xb = x + (size_t)b * C_IN * NHW;
  float* outb = out + (size_t)b * C_IN * NHW;
  for (int i = l; i < QTILE * C_IN; i += 32) {
    int o  = i >> 4;
    int qi = i & 15;
    size_t gidx = (size_t)o * NHW + qwave + qi;
    outb[gidx] = xb[gidx] + sg * ob[o * OPADQ + qi];
  }
}

// ---------------------------------------------------------------------------
extern "C" void kernel_launch(void* const* d_in, const int* in_sizes, int n_in,
                              void* d_out, int out_size, void* d_ws,
                              size_t ws_size, hipStream_t stream) {
  const float* x  = (const float*)d_in[0];
  const float* Wt = (const float*)d_in[1];
  const float* Wp = (const float*)d_in[2];
  const float* Wg = (const float*)d_in[3];
  const float* Wa = (const float*)d_in[4];
  const float* sg = (const float*)d_in[5];
  float* out = (float*)d_out;

  char* ws = (char*)d_ws;
  float*  theta = (float*)ws;                                             // 2 MB
  float*  phi   = (float*)(ws + (size_t)2 * 1024 * 1024);                 // 512 KB
  __bf16* g     = (__bf16*)(ws + (size_t)2 * 1024 * 1024 + 512 * 1024);   // 1 MB

  theta_proj_kernel<<<(16 * NHW) / 256, 256, 0, stream>>>(x, Wt, theta);
  pool_proj_kernel<<<(16 * KP) / 256, 256, 0, stream>>>(x, Wp, Wg, phi, g);
  attn_kernel<<<16 * (NHW / QBLK), 256, 0, stream>>>(x, theta, phi, g, Wa, sg,
                                                     out);
}
